// GNN_72112500899971
// MI455X (gfx1250) — compile-verified, matrix operand order checked by reading, not run on hardware
//
#include <hip/hip_runtime.h>
#include <hip/hip_bf16.h>

#define N_NODES 100000
#define N_EDGES 1600000
#define D 64
#define N_TILES (N_NODES / 16)          // 6250, exact
#define WAVES_PER_BLOCK 2

typedef __attribute__((ext_vector_type(16))) __bf16        v16bf;
typedef __attribute__((ext_vector_type(8)))  float         v8f;
typedef __attribute__((ext_vector_type(8)))  unsigned int  v8u;

// ---- f32 -> bf16 (round to nearest even) -------------------------------
__device__ __forceinline__ unsigned short f32_to_bf16(float f) {
    unsigned int u = __builtin_bit_cast(unsigned int, f);
    u += 0x7FFFu + ((u >> 16) & 1u);
    return (unsigned short)(u >> 16);
}
__device__ __forceinline__ unsigned int pack_bf16x2(float lo, float hi) {
    return (unsigned int)f32_to_bf16(lo) | ((unsigned int)f32_to_bf16(hi) << 16);
}

// ---- utility kernels ---------------------------------------------------
__global__ void zero_kernel(float* p, long n) {
    long i = (long)blockIdx.x * blockDim.x + threadIdx.x;
    if (i < n) p[i] = 0.0f;
}

__global__ void deg_kernel(const long long* __restrict__ ei, float* deg) {
    int e = blockIdx.x * blockDim.x + threadIdx.x;
    if (e < N_EDGES) atomicAdd(&deg[(int)ei[N_EDGES + e]], 1.0f);
}

__global__ void deginv_kernel(float* deg) {
    int n = blockIdx.x * blockDim.x + threadIdx.x;
    if (n < N_NODES) deg[n] = 1.0f / fmaxf(deg[n], 1.0f);
}

__global__ void cvt_w_kernel(const float* __restrict__ w, unsigned short* o, int n) {
    int i = blockIdx.x * blockDim.x + threadIdx.x;
    if (i < n) o[i] = f32_to_bf16(w[i]);
}

// ---- edge scatter: msg[dst,:] += feat[src,:] ---------------------------
// 16 threads per edge, 4 floats each. feat rows hit L2 (matrix is 25.6 MB).
__global__ void scatter_kernel(const float* __restrict__ feat,
                               const long long* __restrict__ ei,
                               float* msg) {
    long t = (long)blockIdx.x * blockDim.x + threadIdx.x;
    if (t >= (long)N_EDGES * 16) return;
    int e = (int)(t >> 4);
    int q = (int)(t & 15);
    int s = (int)ei[e];
    int d = (int)ei[N_EDGES + e];
    const float4 v = *(const float4*)(feat + (long)s * D + q * 4);
    float* dp = msg + (long)d * D + q * 4;
    atomicAdd(dp + 0, v.x);
    atomicAdd(dp + 1, v.y);
    atomicAdd(dp + 2, v.z);
    atomicAdd(dp + 3, v.w);
}

// ---- fused SAGE linear: out = [msg*deg_inv]@Wl^T + feat@Wr^T + b -------
// One wave (32 lanes) per 16-row tile. bf16 WMMA 16x16x32, f32 accumulate.
__global__ __launch_bounds__(WAVES_PER_BLOCK * 32)
void sage_wmma_kernel(const float* feat,
                      const float* __restrict__ msg,
                      const float* __restrict__ deg_inv,
                      const unsigned short* __restrict__ Wl,   // bf16 bits [64][64]
                      const unsigned short* __restrict__ Wr,   // bf16 bits [64][64]
                      const float* __restrict__ bias,
                      float* out, int do_relu)
{
    // [wave][mat: 0=agg 1=feat][row 0..15][col-pair 0..31], padded to 33 to
    // break the stride-32 bank pattern on the 64-bank LDS.
    __shared__ unsigned int lds[WAVES_PER_BLOCK][2][16][33];

    const int lane = threadIdx.x & 31;
    const int wave = threadIdx.x >> 5;
    const int tile = blockIdx.x * WAVES_PER_BLOCK + wave;   // exactly N_TILES
    const long r0  = (long)tile * 16;
    const int  half = lane >> 4;          // 0: lanes 0-15, 1: lanes 16-31
    const int  m    = lane & 15;

    // ---- stage 16 rows of agg and feat into LDS as bf16 pairs ----------
    for (int row = 0; row < 16; ++row) {
        long g = (r0 + row) * D + lane * 2;               // 2 cols per lane
        float2 fm = *(const float2*)(msg  + g);
        float2 fx = *(const float2*)(feat + g);
        float  di = deg_inv[r0 + row];
        lds[wave][0][row][lane] = pack_bf16x2(fm.x * di, fm.y * di);
        lds[wave][1][row][lane] = pack_bf16x2(fx.x, fx.y);
    }
    // per-wave LDS fence (CDNA5 split counters); waves use disjoint slices.
    asm volatile("s_wait_dscnt 0" ::: "memory");

    // ---- A fragments, ISA 16-bit A 16x32 layout ------------------------
    // element e of v16bf: K = (e/8)*16 + half*8 + (e%8), M = lane%16
    v16bf afrag[4];                       // f=0,1: agg K 0/32 ; f=2,3: feat K 0/32
#pragma unroll
    for (int f = 0; f < 4; ++f) {
        const int mat = f >> 1;
        const int kb  = (f & 1) * 32;
        v8u a;
#pragma unroll
        for (int p = 0; p < 8; ++p) {
            int k = kb + (p >> 2) * 16 + half * 8 + (p & 3) * 2;  // even K of pair
            a[p] = lds[wave][mat][m][k >> 1];
        }
        afrag[f] = __builtin_bit_cast(v16bf, a);
    }

    // ---- 4 output column tiles, each accumulating 4 K-steps ------------
    const unsigned int* WlU = (const unsigned int*)Wl;    // bf16 pairs along K
    const unsigned int* WrU = (const unsigned int*)Wr;
#pragma unroll
    for (int nt = 0; nt < 4; ++nt) {
        const int o = nt * 16 + m;        // output feature owned by this lane (B: N=lane%16)
        v8f c = {};
#pragma unroll
        for (int f = 0; f < 4; ++f) {
            const unsigned int* W = (f < 2) ? WlU : WrU;
            const int kb = (f & 1) * 32;
            v8u bb;
#pragma unroll
            for (int p = 0; p < 8; ++p) {
                int k = kb + (p >> 2) * 16 + half * 8 + (p & 3) * 2;
                bb[p] = W[o * 32 + (k >> 1)];             // B[k][n] = W[o][k]
            }
            v16bf bfrag = __builtin_bit_cast(v16bf, bb);
            c = __builtin_amdgcn_wmma_f32_16x16x32_bf16(
                    false, afrag[f], false, bfrag, (short)0, c, false, false);
        }
        const float bv = bias[nt * 16 + m];
#pragma unroll
        for (int j = 0; j < 8; ++j) {     // C layout: M = j + half*8, N = lane%16
            float v = c[j] + bv;
            if (do_relu) v = fmaxf(v, 0.0f);
            out[(r0 + j + half * 8) * D + nt * 16 + m] = v;
        }
    }
}

// ------------------------------------------------------------------------
extern "C" void kernel_launch(void* const* d_in, const int* in_sizes, int n_in,
                              void* d_out, int out_size, void* d_ws, size_t ws_size,
                              hipStream_t stream) {
    (void)in_sizes; (void)n_in; (void)out_size; (void)ws_size;

    const float*     x    = (const float*)d_in[0];
    const long long* ei   = (const long long*)d_in[1];
    const float*     W1_l = (const float*)d_in[2];
    const float*     b1   = (const float*)d_in[3];
    const float*     W1_r = (const float*)d_in[4];
    const float*     W2_l = (const float*)d_in[5];
    const float*     b2   = (const float*)d_in[6];
    const float*     W2_r = (const float*)d_in[7];
    float*           out  = (float*)d_out;       // also reused as h between layers

    // workspace: deg_inv | msg | bf16 weights   (~26 MB)
    float* wsf = (float*)d_ws;
    float* deg = wsf;                            // 100000 (becomes deg_inv)
    float* msg = wsf + 102400;                   // 100000*64
    unsigned short* wbf = (unsigned short*)(wsf + 102400 + (long)N_NODES * D);
    unsigned short *Wl1 = wbf, *Wr1 = wbf + 4096, *Wl2 = wbf + 8192, *Wr2 = wbf + 12288;

    const long MSG_N = (long)N_NODES * D;

    // degree and its inverse (once per launch — deterministic)
    zero_kernel<<<(N_NODES + 255) / 256, 256, 0, stream>>>(deg, N_NODES);
    deg_kernel<<<(N_EDGES + 255) / 256, 256, 0, stream>>>(ei, deg);
    deginv_kernel<<<(N_NODES + 255) / 256, 256, 0, stream>>>(deg);

    // bf16 weight copies
    cvt_w_kernel<<<16, 256, 0, stream>>>(W1_l, Wl1, 4096);
    cvt_w_kernel<<<16, 256, 0, stream>>>(W1_r, Wr1, 4096);
    cvt_w_kernel<<<16, 256, 0, stream>>>(W2_l, Wl2, 4096);
    cvt_w_kernel<<<16, 256, 0, stream>>>(W2_r, Wr2, 4096);

    // ---- layer 1: h = relu(agg(x)@W1_l^T + b1 + x@W1_r^T)  (h -> out) --
    zero_kernel<<<(int)((MSG_N + 255) / 256), 256, 0, stream>>>(msg, MSG_N);
    scatter_kernel<<<(int)(((long)N_EDGES * 16 + 255) / 256), 256, 0, stream>>>(x, ei, msg);
    sage_wmma_kernel<<<N_TILES / WAVES_PER_BLOCK, WAVES_PER_BLOCK * 32, 0, stream>>>(
        x, msg, deg, Wl1, Wr1, b1, out, 1);

    // ---- layer 2: out = agg(h)@W2_l^T + b2 + h@W2_r^T ------------------
    zero_kernel<<<(int)((MSG_N + 255) / 256), 256, 0, stream>>>(msg, MSG_N);
    scatter_kernel<<<(int)(((long)N_EDGES * 16 + 255) / 256), 256, 0, stream>>>(out, ei, msg);
    sage_wmma_kernel<<<N_TILES / WAVES_PER_BLOCK, WAVES_PER_BLOCK * 32, 0, stream>>>(
        out, msg, deg, Wl2, Wr2, b2, out, 0);
}
